// GeneAwareContrastive_15796889715193
// MI455X (gfx1250) — matrix operand order repaired and checked by reading, not compiled, not run on hardware
//
#include <hip/hip_runtime.h>
#include <hip/hip_bf16.h>
#include <stdint.h>

#if __has_include(<hip/amd_detail/amd_gfx1250_TDM.h>)
#define GAC_TDM_6ARG 1
#endif

typedef __attribute__((ext_vector_type(16))) _Float16 v16h;
typedef __attribute__((ext_vector_type(8)))  _Float16 v8h;
typedef __attribute__((ext_vector_type(8)))  float    v8f;
typedef __attribute__((ext_vector_type(4)))  uint32_t u32x4;
typedef __attribute__((ext_vector_type(8)))  int      i32x8;
typedef __attribute__((ext_vector_type(4)))  int      i32x4;

#define TEMP_INV   2.0f   // 1 / TEMPERATURE
#define W_WITHIN   1.0f
#define W_CROSS    0.5f
#define MARGIN     0.1f
#define NBINS      1024

#define DFIX       256            // feature dim (reference-fixed)
#define KSTEPS     (DFIX / 32)    // 8 WMMA k-steps
#define BSTRIDE    264            // padded LDS row stride in halves (528 B)
#define LDS_BYTES  (128 * BSTRIDE * 2)

// ---------------------------------------------------------------- zero init
__global__ void gac_zero(float* rowS, int B, int* hist, int* cursor,
                         float* withinSum, float* crossSum,
                         unsigned long long* crossCnt) {
    int t = blockIdx.x * blockDim.x + threadIdx.x;
    if (t < B) rowS[t] = 0.0f;
    if (t < NBINS) { hist[t] = 0; cursor[t] = 0; }
    if (t == 0) { *withinSum = 0.0f; *crossSum = 0.0f; *crossCnt = 0ull; }
}

// ----------------------------------------------- row L2-normalize + f16 cast
__global__ void gac_normalize(const float* __restrict__ feat,
                              _Float16* __restrict__ fnh, int D) {
    __shared__ float red[256];
    int row = blockIdx.x, t = threadIdx.x;
    float v = feat[(size_t)row * D + t];
    red[t] = v * v;
    __syncthreads();
    for (int s = 128; s > 0; s >>= 1) {
        if (t < s) red[t] += red[t + s];
        __syncthreads();
    }
    float rn = rsqrtf(red[0]);
    fnh[(size_t)row * D + t] = (_Float16)(v * rn);
}

// ---------------------------------------------------------------- histogram
__global__ void gac_hist(const int* __restrict__ labels, int* hist, int B) {
    int t = blockIdx.x * blockDim.x + threadIdx.x;
    if (t < B) atomicAdd(&hist[((unsigned)labels[t]) & (NBINS - 1)], 1);
}

// ------------------------------------------------- exclusive scan, one block
__global__ void gac_scan(const int* __restrict__ hist, int* offs) {
    __shared__ int s[NBINS];
    int t = threadIdx.x;
    s[t] = hist[t];
    __syncthreads();
    for (int d = 1; d < NBINS; d <<= 1) {
        int x = (t >= d) ? s[t - d] : 0;
        __syncthreads();
        s[t] += x;
        __syncthreads();
    }
    offs[t] = s[t] - hist[t];   // exclusive
}

// ----------------------------------------------------------------- scatter
__global__ void gac_scatter(const int* __restrict__ labels, const int* offs,
                            int* cursor, int* members, int B) {
    int t = blockIdx.x * blockDim.x + threadIdx.x;
    if (t < B) {
        unsigned g = ((unsigned)labels[t]) & (NBINS - 1);
        int pos = atomicAdd(&cursor[g], 1);
        members[offs[g] + pos] = t;
    }
}

// -------------------------------------------------------- main WMMA pass
// grid (B/128, B/128), block 256 (8 wave32). Each wave: 16 rows x 128 cols.
// B panel staged into LDS by the Tensor Data Mover. Per k-step: batch-load
// all 8 B fragments (16 clustered ds_load_b128), then 8 WMMAs against them.
__global__ void __launch_bounds__(256)
gac_pass1(const _Float16* __restrict__ fnh, const int* __restrict__ labels,
          float* __restrict__ rowS, float* crossSum,
          unsigned long long* crossCnt, int Btot) {
    extern __shared__ _Float16 Bs[];          // [128][BSTRIDE]
    const int tid  = threadIdx.x;
    const int lane = tid & 31;
    const int wave = tid >> 5;
    const int l    = lane & 15;
    const int hi   = lane >> 4;
    const int rowBase = blockIdx.y * 128 + wave * 16;
    const int colBase = blockIdx.x * 128;

    const _Float16* arow = fnh + (size_t)(rowBase + l) * DFIX;

    auto loadA = [&](int ks) -> v16h {        // ISA 16-bit A 16x32 layout
        const int k0 = ks * 32 + (hi ? 8 : 0);
        v8h x = *(const v8h*)(arow + k0);
        v8h y = *(const v8h*)(arow + k0 + 16);
        v16h a;
#pragma unroll
        for (int e = 0; e < 8; ++e) { a[e] = x[e]; a[8 + e] = y[e]; }
        return a;
    };
    auto loadB = [&](int ks, int t) -> v16h { // ISA 16-bit B KxN layout
        const _Float16* bp = &Bs[(t * 16 + l) * BSTRIDE + ks * 32 + (hi ? 8 : 0)];
        v8h x = *(const v8h*)bp;
        v8h y = *(const v8h*)(bp + 16);
        v16h b;
#pragma unroll
        for (int e = 0; e < 8; ++e) { b[e] = x[e]; b[8 + e] = y[e]; }
        return b;
    };

    // ---- stage B panel (128 cols x 256 K halves) into LDS via TDM ----
#if __has_builtin(__builtin_amdgcn_tensor_load_to_lds)
    if (tid < 32) {                    // one wave issues the TDM transfer
        const uint64_t ga =
            (uint64_t)(uintptr_t)fnh + (uint64_t)colBase * (DFIX * 2);
        const uint32_t ldsOff = (uint32_t)(uintptr_t)&Bs[0];
        // D# group 0: count=1, lds_addr, global_addr, type=2
        u32x4 g0;
        g0[0] = 1u;
        g0[1] = ldsOff;
        g0[2] = (uint32_t)ga;
        g0[3] = (uint32_t)(ga >> 32) | (2u << 30);
        // D# group 1: data_size=3 (8B units), pad_enable, pad_interval=6
        // (512B), pad_amount=3 (16B) -> 528B LDS row stride
        const uint32_t rowU = (DFIX * 2) / 8;          // 64 units per row
        i32x8 g1;
        g1[0] = (int)((3u << 16) | (1u << 20) | (6u << 22) | (3u << 25));
        g1[1] = (int)((rowU & 0xffffu) << 16);          // tensor_dim0 lo
        g1[2] = (int)(((rowU >> 16) & 0xffffu) |        // tensor_dim0 hi
                      (((uint32_t)Btot & 0xffffu) << 16)); // tensor_dim1 lo
        g1[3] = (int)((((uint32_t)Btot >> 16) & 0xffffu) |
                      ((rowU & 0xffffu) << 16));        // tile_dim0 = rowU
        g1[4] = 128;                                    // tile_dim1 = 128 rows
        g1[5] = (int)rowU;                              // tensor_dim0_stride
        g1[6] = 0;
        g1[7] = 0;
        i32x4 z4 = {};
#ifdef GAC_TDM_6ARG
        i32x8 z8 = {};
        __builtin_amdgcn_tensor_load_to_lds(g0, g1, z4, z4, z8, 0);
#else
        __builtin_amdgcn_tensor_load_to_lds(g0, g1, z4, z4, 0);
#endif
        __builtin_amdgcn_s_wait_tensorcnt(0);
    }
    __syncthreads();
#else
    // fallback: cooperative fill via ds_store_b128
#pragma unroll
    for (int it = 0; it < 16; ++it) {
        int idx = it * 256 + tid;               // 0..4095 chunks of 16B
        int row = idx >> 5;
        int c   = (idx & 31) * 8;
        v8h v = *(const v8h*)(fnh + (size_t)(colBase + row) * DFIX + c);
        *(v8h*)(&Bs[row * BSTRIDE + c]) = v;
    }
    __syncthreads();
#endif

    // ---- 64 straight-line WMMAs: per k-step, batch 16 ds loads then 8 MMAs
    v8f acc[8];
#pragma unroll
    for (int t = 0; t < 8; ++t) { v8f z = {}; acc[t] = z; }

    v16h aCur = loadA(0);
#pragma unroll
    for (int ks = 0; ks < KSTEPS; ++ks) {
        v16h b[8];
#pragma unroll
        for (int t = 0; t < 8; ++t) b[t] = loadB(ks, t);   // cluster DS loads
        v16h aNext = aCur;
        if (ks + 1 < KSTEPS) aNext = loadA(ks + 1);        // distance-8 global
#pragma unroll
        for (int t = 0; t < 8; ++t)
            acc[t] = __builtin_amdgcn_wmma_f32_16x16x32_f16(
                false, aCur, false, b[t], (short)0, acc[t], false, false);
        aCur = aNext;
    }

    // ---- epilogue: consume sim tile in registers ----
    const int ibase = rowBase + hi * 8;
    int li[8];
#pragma unroll
    for (int r = 0; r < 8; ++r) li[r] = labels[ibase + r];

    float rowExp[8];
#pragma unroll
    for (int r = 0; r < 8; ++r) rowExp[r] = 0.0f;
    float cs = 0.0f;
    unsigned nc = 0;

#pragma unroll
    for (int t = 0; t < 8; ++t) {
        int lj = labels[colBase + t * 16 + l];
#pragma unroll
        for (int r = 0; r < 8; ++r) {
            float s = acc[t][r] * TEMP_INV;
            bool neg = (li[r] != lj);          // diagonal excluded (same label)
            float e = __expf(s);
            float h = fmaxf(s - MARGIN, 0.0f);
            if (neg) { rowExp[r] += e; cs += h; nc++; }
        }
    }

    // per-row sums: reduce across the 16 lanes of each half-wave
#pragma unroll
    for (int off = 1; off < 16; off <<= 1)
#pragma unroll
        for (int r = 0; r < 8; ++r)
            rowExp[r] += __shfl_xor(rowExp[r], off, 32);
    if (l == 0) {
#pragma unroll
        for (int r = 0; r < 8; ++r)
            atomicAdd(&rowS[ibase + r], rowExp[r]);
    }

    // cross sums: reduce across full wave
    float ncf = (float)nc;
#pragma unroll
    for (int off = 1; off < 32; off <<= 1) {
        cs  += __shfl_xor(cs, off, 32);
        ncf += __shfl_xor(ncf, off, 32);
    }
    if (lane == 0) {
        atomicAdd(crossSum, cs);
        atomicAdd(crossCnt, (unsigned long long)(ncf + 0.5f));
    }
}

// ---------------------------------------------- within-gene loss (sparse)
__global__ void gac_within(const _Float16* __restrict__ fnh,
                           const int* __restrict__ labels,
                           const int* __restrict__ cnt, const int* __restrict__ offs,
                           const int* __restrict__ members,
                           const float* __restrict__ rowS,
                           float* withinSum, int D) {
    __shared__ float red[256];
    int g = blockIdx.x;
    int k = cnt[g];
    float acc = 0.0f;
    if (k >= 2) {
        int off = offs[g];
        if (labels[members[off]] >= 0) {            // gene -1 excluded
            int kk = k * k;
            for (int idx = threadIdx.x; idx < kk; idx += blockDim.x) {
                int a = idx / k, b = idx - a * k;
                if (b <= a) continue;
                int pa = members[off + a], pb = members[off + b];
                int i = min(pa, pb), j = max(pa, pb); // anchor = smaller index
                float Si = rowS[i];
                if (Si <= 0.0f) continue;             // anchor has no negatives
                const v8h* x = (const v8h*)(fnh + (size_t)i * D);
                const v8h* y = (const v8h*)(fnh + (size_t)j * D);
                float d = 0.0f;
                for (int c = 0; c < D / 8; ++c) {
                    v8h xv = x[c], yv = y[c];
#pragma unroll
                    for (int e = 0; e < 8; ++e)
                        d += (float)xv[e] * (float)yv[e];
                }
                float sim = d * TEMP_INV;
                float arg = __logf(Si) - sim;         // lse_neg[i] - sim_ij
                acc += (arg > 15.0f) ? arg : log1pf(__expf(arg));
            }
        }
    }
    red[threadIdx.x] = acc;
    __syncthreads();
    for (int s = 128; s > 0; s >>= 1) {
        if (threadIdx.x < s) red[threadIdx.x] += red[threadIdx.x + s];
        __syncthreads();
    }
    if (threadIdx.x == 0 && red[0] != 0.0f) atomicAdd(withinSum, red[0]);
}

// ------------------------------------------------------------------ finalize
__global__ void gac_final(const int* __restrict__ cnt, const float* withinSum,
                          const float* crossSum, const unsigned long long* crossCnt,
                          float* out) {
    if (blockIdx.x == 0 && threadIdx.x == 0) {
        long long nw = 0;
        for (int g = 0; g < NBINS; ++g) {
            long long c = cnt[g];
            nw += c * (c - 1);
        }
        float within = *withinSum;
        unsigned long long cc = *crossCnt;
        float cross = (cc > 0ull) ? (*crossSum) / (float)cc : 0.0f;
        out[0] = W_WITHIN * within + W_CROSS * cross;
        out[1] = within;
        out[2] = cross;
        out[3] = (float)nw;
        out[4] = (float)cc;
    }
}

extern "C" void kernel_launch(void* const* d_in, const int* in_sizes, int n_in,
                              void* d_out, int out_size, void* d_ws, size_t ws_size,
                              hipStream_t stream) {
    const float* feat  = (const float*)d_in[0];
    const int* labels  = (const int*)d_in[1];
    float* out         = (float*)d_out;

    const int B = in_sizes[1];                 // 8192
    const int D = in_sizes[0] / B;             // 256 (pass1 assumes this)

    // ---- workspace carve-up (~4.3 MB) ----
    char* p = (char*)d_ws;
    _Float16* fnh = (_Float16*)p;              p += (size_t)B * D * sizeof(_Float16);
    float* rowS   = (float*)p;                 p += (size_t)B * sizeof(float);
    int* hist     = (int*)p;                   p += NBINS * sizeof(int);
    int* offs     = (int*)p;                   p += NBINS * sizeof(int);
    int* cursor   = (int*)p;                   p += NBINS * sizeof(int);
    int* members  = (int*)p;                   p += (size_t)B * sizeof(int);
    p = (char*)(((uintptr_t)p + 7) & ~(uintptr_t)7);
    unsigned long long* crossCnt = (unsigned long long*)p; p += 8;
    float* withinSum = (float*)p;              p += 4;
    float* crossSum  = (float*)p;              p += 4;
    (void)ws_size; (void)n_in; (void)out_size;

    const int zBlocks = (B + 255) / 256;
    gac_zero<<<zBlocks, 256, 0, stream>>>(rowS, B, hist, cursor,
                                          withinSum, crossSum, crossCnt);
    gac_normalize<<<B, D, 0, stream>>>(feat, fnh, D);
    gac_hist<<<zBlocks, 256, 0, stream>>>(labels, hist, B);
    gac_scan<<<1, NBINS, 0, stream>>>(hist, offs);
    gac_scatter<<<zBlocks, 256, 0, stream>>>(labels, offs, cursor, members, B);

    dim3 grid(B / 128, B / 128);
    gac_pass1<<<grid, 256, LDS_BYTES, stream>>>(fnh, labels, rowS,
                                                crossSum, crossCnt, B);
    gac_within<<<NBINS, 256, 0, stream>>>(fnh, labels, hist, offs, members,
                                          rowS, withinSum, D);
    gac_final<<<1, 1, 0, stream>>>(hist, withinSum, crossSum, crossCnt, out);
}
